// MultiStepGRUDecoder_120259084461
// MI455X (gfx1250) — matrix operand order repaired and checked by reading, not compiled, not run on hardware
//
#include <hip/hip_runtime.h>
#include <hip/hip_bf16.h>

// ---------------------------------------------------------------------------
// MultiStepGRUDecoder for MI455X (gfx1250, wave32, WMMA bf16)
//
// Sizes: H=512, B=64, S=512, N_STEPS=24.
// All GEMMs use v_wmma_f32_16x16x32_bf16 (bf16 inputs, fp32 accumulate),
// register-blocked 1 M-tile x 4 N-tiles per wave for A-fragment reuse.
// enc_outputs and enc_proj are kept in bf16 so both (32MB each) live in the
// 192MB L2 across the 24 sequential decode steps; pointwise math is fp32.
// ---------------------------------------------------------------------------

typedef __bf16 bf16_t;
typedef __bf16 v16bf __attribute__((ext_vector_type(16)));
typedef __bf16 v8bf  __attribute__((ext_vector_type(8)));
typedef float  v8f   __attribute__((ext_vector_type(8)));

#define HID   512
#define BATCH 64
#define SEQ   512
#define NSTEP 24

__device__ __forceinline__ float fast_tanh(float x) {
    // tanh(x) = 1 - 2/(exp(2x)+1); saturates correctly for |x| large
    float e = __expf(2.0f * x);
    return 1.0f - 2.0f / (e + 1.0f);
}
__device__ __forceinline__ float fast_sigmoid(float x) {
    return 1.0f / (1.0f + __expf(-x));
}
__device__ __forceinline__ float wave_reduce_sum(float v) {
#pragma unroll
    for (int m = 16; m >= 1; m >>= 1) v += __shfl_xor(v, m, 32);
    return v;
}

// ---------------------------------------------------------------------------
// Conversions
// ---------------------------------------------------------------------------
__global__ void cvt_bf16(const float* __restrict__ in, bf16_t* __restrict__ out, int n) {
    int i = blockIdx.x * blockDim.x + threadIdx.x;
    if (i < n) out[i] = (bf16_t)in[i];
}

// in[K][N] row-major  ->  out[N][K] row-major (B-fragment friendly)
__global__ void cvt_t_bf16(const float* __restrict__ in, bf16_t* __restrict__ out, int K, int N) {
    int i = blockIdx.x * blockDim.x + threadIdx.x;
    if (i < K * N) {
        int n = i / K, k = i % K;
        out[i] = (bf16_t)in[(size_t)k * N + n];
    }
}

__global__ void init_h(const float* __restrict__ h0, float* __restrict__ h_f32,
                       bf16_t* __restrict__ h_bf, bf16_t* __restrict__ x_bf) {
    int i = blockIdx.x * blockDim.x + threadIdx.x;   // 0 .. B*H-1
    if (i < BATCH * HID) {
        float v = h0[i];
        h_f32[i] = v;
        h_bf[i]  = (bf16_t)v;
        int b = i >> 9, j = i & (HID - 1);
        x_bf[b * (2 * HID) + HID + j] = (bf16_t)v;   // h-half of [context; h]
    }
}

__global__ void copy_f32(const float* __restrict__ src, float* __restrict__ dst, int n) {
    int i = blockIdx.x * blockDim.x + threadIdx.x;
    if (i < n) dst[i] = src[i];
}

// ---------------------------------------------------------------------------
// WMMA GEMM, register blocked: each wave computes one 16x64 output block
// (1 M-tile x 4 N-tiles), reusing the A fragment across 4 WMMAs per K-step.
// C[M,N] = A[M,K](bf16,row) * Wt[N,K](bf16,row == B^T) (+ bias[n]) (+ ReLU).
// Fragment layouts used here (wave32; A and B share contiguous k order):
//   A 16x32 bf16 : lane half selects K base (0/16); elems 0-15 -> k..k+15
//                  contiguous  => two contiguous 16B loads
//   B 32x16 bf16 : lane n = lane&15, K = (lane>>4)*16 + 0..15 contiguous
//   C/D 16x16 f32: lane n = lane&15, rows m = (lane>>4)*8 + r
// ---------------------------------------------------------------------------
template <bool RELU, bool WRITE_F32, bool WRITE_BF16>
__global__ void wmma_gemm(const bf16_t* __restrict__ A,
                          const bf16_t* __restrict__ Wt,
                          const float* __restrict__ bias,
                          float* __restrict__ C,
                          bf16_t* __restrict__ Cbf,
                          int M, int N, int K) {
    const int wave   = blockIdx.x * (blockDim.x >> 5) + (threadIdx.x >> 5);
    const int ngrp   = N >> 6;                    // groups of 4 N-tiles
    const int nblk   = (M >> 4) * ngrp;
    if (wave >= nblk) return;
    const int tm = wave / ngrp;
    const int tg = wave % ngrp;                   // n base = tg*64

    const int lane  = threadIdx.x & 31;
    const int lhalf = lane >> 4;                  // 0 | 1
    const int lmod  = lane & 15;

    const bf16_t* pA = A  + (size_t)(tm * 16 + lmod) * K + lhalf * 16;
    const bf16_t* pB = Wt + (size_t)(tg * 64 + lmod) * K + lhalf * 16;
    const size_t bstride = (size_t)16 * K;        // next N-tile in Wt

    v8f acc[4] = {};
    for (int k = 0; k < K; k += 32) {
        v8bf alo = *(const v8bf*)(pA + k);
        v8bf ahi = *(const v8bf*)(pA + k + 8);
        v16bf a;
#pragma unroll
        for (int i = 0; i < 8; ++i) { a[i] = alo[i]; a[8 + i] = ahi[i]; }
#pragma unroll
        for (int nb = 0; nb < 4; ++nb) {
            v16bf b = *(const v16bf*)(pB + (size_t)nb * bstride + k);
            acc[nb] = __builtin_amdgcn_wmma_f32_16x16x32_bf16(
                /*neg_a=*/false, a, /*neg_b=*/false, b,
                /*c_mod=*/(short)0, acc[nb], /*reuse_a=*/false, /*reuse_b=*/false);
        }
    }

#pragma unroll
    for (int nb = 0; nb < 4; ++nb) {
        const int n  = tg * 64 + nb * 16 + lmod;
        const float bv = bias ? bias[n] : 0.0f;
#pragma unroll
        for (int r = 0; r < 8; ++r) {
            int m = tm * 16 + lhalf * 8 + r;
            float v = acc[nb][r] + bv;
            if (RELU) v = v > 0.0f ? v : 0.0f;
            if (WRITE_F32)  C[(size_t)m * N + n]   = v;
            if (WRITE_BF16) Cbf[(size_t)m * N + n] = (bf16_t)v;
        }
    }
}

// ---------------------------------------------------------------------------
// scores[b,s] = sum_h tanh(enc_proj[b,s,h] + q[b,h]) * v[h]   (one wave / row)
// Each lane owns a contiguous 16-element chunk: 32B bf16 + vector f32 loads.
// ---------------------------------------------------------------------------
__global__ void attn_score(const bf16_t* __restrict__ enc_proj,
                           const float* __restrict__ q,
                           const float* __restrict__ v,
                           float* __restrict__ scores) {
    int row  = blockIdx.x * (blockDim.x >> 5) + (threadIdx.x >> 5);  // b*S + s
    int lane = threadIdx.x & 31;
    int b = row >> 9;                                                 // S = 512
    const int c = lane * 16;
    const bf16_t* ep = enc_proj + (size_t)row * HID + c;
    const float*  qb = q + b * HID + c;
    const float*  vb = v + c;

    v16bf e = *(const v16bf*)ep;
    float sum = 0.0f;
#pragma unroll
    for (int i = 0; i < 16; ++i) {
        float x = (float)e[i] + qb[i];
        sum += fast_tanh(x) * vb[i];
    }
    sum = wave_reduce_sum(sum);
    if (lane == 0) scores[row] = sum;
}

// softmax over S=512 per batch row; one block of 256 per b
__global__ void softmax512(const float* __restrict__ scores, float* __restrict__ alpha) {
    __shared__ float red[256];
    int b = blockIdx.x, t = threadIdx.x;
    const float* s = scores + b * SEQ;
    float x0 = s[t], x1 = s[t + 256];
    red[t] = fmaxf(x0, x1);
    __syncthreads();
    for (int o = 128; o > 0; o >>= 1) { if (t < o) red[t] = fmaxf(red[t], red[t + o]); __syncthreads(); }
    float m = red[0];
    __syncthreads();
    float e0 = __expf(x0 - m), e1 = __expf(x1 - m);
    red[t] = e0 + e1;
    __syncthreads();
    for (int o = 128; o > 0; o >>= 1) { if (t < o) red[t] += red[t + o]; __syncthreads(); }
    float inv = 1.0f / red[0];
    alpha[b * SEQ + t]       = e0 * inv;
    alpha[b * SEQ + t + 256] = e1 * inv;
}

// context[b,h] = sum_s alpha[b,s] * enc[b,s,h]; also writes bf16 into x=[ctx;h]
__global__ void context_kernel(const float* __restrict__ alpha,
                               const bf16_t* __restrict__ enc,
                               float* __restrict__ ctx,
                               bf16_t* __restrict__ x_bf) {
    int b = blockIdx.x >> 1;
    int h = ((blockIdx.x & 1) << 8) + threadIdx.x;
    const float*  al = alpha + b * SEQ;
    const bf16_t* e  = enc + (size_t)b * SEQ * HID + h;
    float acc = 0.0f;
#pragma unroll 8
    for (int s = 0; s < SEQ; ++s) acc += al[s] * (float)e[(size_t)s * HID];
    ctx[b * HID + h]        = acc;
    x_bf[b * (2 * HID) + h] = (bf16_t)acc;
}

// GRU pointwise + LayerNorm, fused; one block (256 thr) per batch row.
// gi/gh already include b_ih/b_hh from the GEMMs.
__global__ void gru_ln(const float* __restrict__ gi, const float* __restrict__ gh,
                       const float* __restrict__ ln_g, const float* __restrict__ ln_b,
                       float* __restrict__ h_f32, bf16_t* __restrict__ h_bf,
                       bf16_t* __restrict__ x_bf, bf16_t* __restrict__ y_bf) {
    __shared__ float red1[256];
    __shared__ float red2[256];
    int b = blockIdx.x, t = threadIdx.x;
    const float* gib = gi + b * 3 * HID;
    const float* ghb = gh + b * 3 * HID;
    float hnew[2];
    float s1 = 0.0f, s2 = 0.0f;
#pragma unroll
    for (int p = 0; p < 2; ++p) {
        int j = t + p * 256;
        float r = fast_sigmoid(gib[j] + ghb[j]);
        float z = fast_sigmoid(gib[HID + j] + ghb[HID + j]);
        float n = fast_tanh(gib[2 * HID + j] + r * ghb[2 * HID + j]);
        float hp = h_f32[b * HID + j];
        float hv = (1.0f - z) * n + z * hp;
        hnew[p] = hv;
        s1 += hv;
        s2 += hv * hv;
    }
    red1[t] = s1; red2[t] = s2;
    __syncthreads();
    for (int o = 128; o > 0; o >>= 1) {
        if (t < o) { red1[t] += red1[t + o]; red2[t] += red2[t + o]; }
        __syncthreads();
    }
    float mu  = red1[0] * (1.0f / HID);
    float var = red2[0] * (1.0f / HID) - mu * mu;
    float inv = rsqrtf(var + 1e-5f);
#pragma unroll
    for (int p = 0; p < 2; ++p) {
        int j = t + p * 256;
        float hv = hnew[p];
        h_f32[b * HID + j] = hv;
        h_bf[b * HID + j]  = (bf16_t)hv;
        x_bf[b * (2 * HID) + HID + j] = (bf16_t)hv;
        float y = (hv - mu) * inv * ln_g[j] + ln_b[j];
        y_bf[b * HID + j] = (bf16_t)y;
    }
}

// p[b] = y1[b,:] . W2 + b2   -> out[b*NSTEP + t]; one wave per b
__global__ void head2(const float* __restrict__ y1, const float* __restrict__ W2,
                      const float* __restrict__ b2, float* __restrict__ out, int t_step) {
    int b    = blockIdx.x * (blockDim.x >> 5) + (threadIdx.x >> 5);
    int lane = threadIdx.x & 31;
    if (b >= BATCH) return;
    const float* y = y1 + b * (HID / 2);
    float sum = 0.0f;
#pragma unroll
    for (int i = lane; i < HID / 2; i += 32) sum += y[i] * W2[i];
    sum = wave_reduce_sum(sum);
    if (lane == 0) out[b * NSTEP + t_step] = sum + b2[0];
}

// ---------------------------------------------------------------------------
extern "C" void kernel_launch(void* const* d_in, const int* in_sizes, int n_in,
                              void* d_out, int out_size, void* d_ws, size_t ws_size,
                              hipStream_t stream) {
    (void)in_sizes; (void)n_in; (void)out_size; (void)ws_size;

    const float* enc    = (const float*)d_in[0];   // (B,S,H)
    const float* h0     = (const float*)d_in[1];   // (1,B,H)
    const float* We     = (const float*)d_in[2];   // (H,H)
    const float* Wd     = (const float*)d_in[3];   // (H,H)
    const float* attn_b = (const float*)d_in[4];   // (H,)
    const float* attn_v = (const float*)d_in[5];   // (H,)
    const float* W_ih   = (const float*)d_in[6];   // (3H,2H) row-major == Wt layout
    const float* W_hh   = (const float*)d_in[7];   // (3H,H)  row-major == Wt layout
    const float* b_ih   = (const float*)d_in[8];
    const float* b_hh   = (const float*)d_in[9];
    const float* ln_g   = (const float*)d_in[10];
    const float* ln_b   = (const float*)d_in[11];
    const float* W1     = (const float*)d_in[12];  // (H, H/2)
    const float* b1     = (const float*)d_in[13];
    const float* W2     = (const float*)d_in[14];  // (H/2, 1)
    const float* b2     = (const float*)d_in[15];
    float* out = (float*)d_out;                    // [B*NSTEP preds][B*H h_final]

    // ---- workspace layout ----
    char*  ws  = (char*)d_ws;
    size_t off = 0;
    auto alloc = [&](size_t bytes) -> char* {
        char* p = ws + off;
        off += (bytes + 255) & ~(size_t)255;
        return p;
    };
    const size_t BSH = (size_t)BATCH * SEQ * HID;
    bf16_t* enc_bf     = (bf16_t*)alloc(2 * BSH);
    bf16_t* enc_proj   = (bf16_t*)alloc(2 * BSH);
    bf16_t* Wet        = (bf16_t*)alloc(2 * HID * HID);
    bf16_t* Wdt        = (bf16_t*)alloc(2 * HID * HID);
    bf16_t* Wih_bf     = (bf16_t*)alloc(2 * 3 * HID * 2 * HID);
    bf16_t* Whh_bf     = (bf16_t*)alloc(2 * 3 * HID * HID);
    bf16_t* W1t        = (bf16_t*)alloc(2 * (HID / 2) * HID);
    float*  q_buf      = (float*)alloc(4 * BATCH * HID);
    float*  scores     = (float*)alloc(4 * BATCH * SEQ);
    float*  alpha      = (float*)alloc(4 * BATCH * SEQ);
    float*  ctx        = (float*)alloc(4 * BATCH * HID);
    float*  gi         = (float*)alloc(4 * BATCH * 3 * HID);
    float*  gh         = (float*)alloc(4 * BATCH * 3 * HID);
    float*  h_f32      = (float*)alloc(4 * BATCH * HID);
    bf16_t* h_bf       = (bf16_t*)alloc(2 * BATCH * HID);
    bf16_t* x_bf       = (bf16_t*)alloc(2 * BATCH * 2 * HID);
    bf16_t* y_bf       = (bf16_t*)alloc(2 * BATCH * HID);
    float*  y1         = (float*)alloc(4 * BATCH * (HID / 2));

    const dim3 T(256);

    // ---- one-time setup (re-done each call: deterministic) ----
    cvt_bf16<<<(int)((BSH + 255) / 256), T, 0, stream>>>(enc, enc_bf, (int)BSH);
    cvt_t_bf16<<<(HID * HID + 255) / 256, T, 0, stream>>>(We, Wet, HID, HID);
    cvt_t_bf16<<<(HID * HID + 255) / 256, T, 0, stream>>>(Wd, Wdt, HID, HID);
    cvt_bf16<<<(3 * HID * 2 * HID + 255) / 256, T, 0, stream>>>(W_ih, Wih_bf, 3 * HID * 2 * HID);
    cvt_bf16<<<(3 * HID * HID + 255) / 256, T, 0, stream>>>(W_hh, Whh_bf, 3 * HID * HID);
    cvt_t_bf16<<<(HID * (HID / 2) + 255) / 256, T, 0, stream>>>(W1, W1t, HID, HID / 2);
    init_h<<<(BATCH * HID + 255) / 256, T, 0, stream>>>(h0, h_f32, h_bf, x_bf);

    // enc_proj = enc @ We  (M=B*S=32768, N=512, K=512) -> bf16
    {
        int blocks16x64 = (BATCH * SEQ / 16) * (HID / 64);     // 16384 waves
        wmma_gemm<false, false, true><<<(blocks16x64 + 7) / 8, T, 0, stream>>>(
            enc_bf, Wet, nullptr, nullptr, enc_proj, BATCH * SEQ, HID, HID);
    }

    // ---- 24 sequential decode steps ----
    for (int t = 0; t < NSTEP; ++t) {
        // q = h @ Wd + attn_b   (64 x 512 x 512)
        {
            int blk = (BATCH / 16) * (HID / 64);               // 32 waves
            wmma_gemm<false, true, false><<<(blk + 7) / 8, T, 0, stream>>>(
                h_bf, Wdt, attn_b, q_buf, nullptr, BATCH, HID, HID);
        }
        // scores: one wave per (b,s)
        attn_score<<<BATCH * SEQ / 8, T, 0, stream>>>(enc_proj, q_buf, attn_v, scores);
        // softmax over s
        softmax512<<<BATCH, T, 0, stream>>>(scores, alpha);
        // context = alpha . enc ; also fills x[:, :H]
        context_kernel<<<BATCH * 2, T, 0, stream>>>(alpha, enc_bf, ctx, x_bf);
        // gi = [ctx;h] @ W_ih^T + b_ih   (64 x 1536 x 1024)
        {
            int blk = (BATCH / 16) * (3 * HID / 64);           // 96 waves
            wmma_gemm<false, true, false><<<(blk + 7) / 8, T, 0, stream>>>(
                x_bf, Wih_bf, b_ih, gi, nullptr, BATCH, 3 * HID, 2 * HID);
        }
        // gh = h @ W_hh^T + b_hh         (64 x 1536 x 512)
        {
            int blk = (BATCH / 16) * (3 * HID / 64);           // 96 waves
            wmma_gemm<false, true, false><<<(blk + 7) / 8, T, 0, stream>>>(
                h_bf, Whh_bf, b_hh, gh, nullptr, BATCH, 3 * HID, HID);
        }
        // fused GRU gates + LayerNorm; updates h (f32/bf16/x) and y_bf
        gru_ln<<<BATCH, T, 0, stream>>>(gi, gh, ln_g, ln_b, h_f32, h_bf, x_bf, y_bf);
        // y1 = relu(y @ W1 + b1)         (64 x 256 x 512)
        {
            int blk = (BATCH / 16) * ((HID / 2) / 64);         // 16 waves
            wmma_gemm<true, true, false><<<(blk + 7) / 8, T, 0, stream>>>(
                y_bf, W1t, b1, y1, nullptr, BATCH, HID / 2, HID);
        }
        // p = y1 @ W2 + b2 -> out[b*NSTEP + t]
        head2<<<(BATCH + 7) / 8, T, 0, stream>>>(y1, W2, b2, out, t);
    }

    // h_final -> out tail
    copy_f32<<<(BATCH * HID + 255) / 256, T, 0, stream>>>(h_f32, out + BATCH * NSTEP, BATCH * HID);
}
